// RoutingTransformerEncoder_35467839930472
// MI455X (gfx1250) — matrix operand order repaired and checked by reading, not compiled, not run on hardware
//
#include <hip/hip_runtime.h>
#include <cmath>

typedef __attribute__((ext_vector_type(16))) __bf16 bf16x16;
typedef __attribute__((ext_vector_type(8)))  __bf16 bf16x8;
typedef __attribute__((ext_vector_type(8)))  float  f32x8;

namespace cfg {
constexpr int B  = 2048, S = 20, D = 512, H = 8, DH = 64, L = 12, FF = 2048;
constexpr int M  = B * S;          // 40960 token rows
constexpr float SCALE = 0.125f;    // DH^-0.5
constexpr float NEG   = -1e9f;
}

// ---------------- workspace layout (bytes) ----------------
constexpr size_t OFF_H   = 0;                                           // f32  M x D
constexpr size_t OFF_Y   = OFF_H   + (size_t)cfg::M * cfg::D  * 4;      // bf16 M x D
constexpr size_t OFF_QKV = OFF_Y   + (size_t)cfg::M * cfg::D  * 2;      // bf16 M x 3D
constexpr size_t OFF_ATT = OFF_QKV + (size_t)cfg::M * 3*cfg::D * 2;     // bf16 M x D
constexpr size_t OFF_FF  = OFF_ATT + (size_t)cfg::M * cfg::D  * 2;      // bf16 M x FF
constexpr size_t OFF_WQ  = OFF_FF  + (size_t)cfg::M * cfg::FF * 2;      // bf16 L x (3D x D)  (transposed)
constexpr size_t OFF_WO  = OFF_WQ  + (size_t)cfg::L * cfg::D * 3*cfg::D * 2;
constexpr size_t OFF_W1  = OFF_WO  + (size_t)cfg::L * cfg::D * cfg::D  * 2;
constexpr size_t OFF_W2  = OFF_W1  + (size_t)cfg::L * cfg::D * cfg::FF * 2;

// ---------------- async global -> LDS copy (CDNA5, ASYNCcnt) ----------------
__device__ inline void async_b128(unsigned lds_off, const void* gptr) {
  asm volatile("global_load_async_to_lds_b128 %0, %1, off"
               :: "v"(lds_off), "v"(gptr) : "memory");
}
__device__ inline void wait_async0() {
  asm volatile("s_wait_asynccnt 0x0" ::: "memory");
}

// ---------------- small utility kernels ----------------
__global__ void k_f32_to_bf16(const float* __restrict__ src, __bf16* __restrict__ dst, int n) {
  int i = blockIdx.x * blockDim.x + threadIdx.x;
  if (i < n) dst[i] = (__bf16)src[i];
}

// transpose one layer's (K x N) f32 weight into (N x K) bf16; blockIdx.z = layer
__global__ __launch_bounds__(256)
void k_transpose_bf16(const float* __restrict__ src, __bf16* __restrict__ dst, int K, int N) {
  __shared__ float t[32][33];
  const float* s = src + (size_t)blockIdx.z * K * N;
  __bf16*      d = dst + (size_t)blockIdx.z * K * N;
  int k0 = blockIdx.y * 32, n0 = blockIdx.x * 32;
  int tx = threadIdx.x & 31, ty = threadIdx.x >> 5;      // 32 x 8
#pragma unroll
  for (int r = 0; r < 4; ++r)
    t[ty + r * 8][tx] = s[(size_t)(k0 + ty + r * 8) * N + n0 + tx];
  __syncthreads();
#pragma unroll
  for (int r = 0; r < 4; ++r)
    d[(size_t)(n0 + ty + r * 8) * K + k0 + tx] = (__bf16)t[tx][ty + r * 8];
}

__global__ void k_embed(const int* __restrict__ x, const float* __restrict__ tok,
                        const float* __restrict__ pos, float* __restrict__ h) {
  int row = blockIdx.x;
  int t   = x[row];
  int s   = row % cfg::S;
  for (int d = threadIdx.x; d < cfg::D; d += blockDim.x)
    h[(size_t)row * cfg::D + d] = tok[(size_t)t * cfg::D + d] + pos[(size_t)s * cfg::D + d];
}

// one wave per row, lane handles 16 contiguous floats
__global__ __launch_bounds__(256)
void k_ln_bf16(const float* __restrict__ h, const float* __restrict__ sc,
               const float* __restrict__ bi, __bf16* __restrict__ y, int nrows) {
  int row  = blockIdx.x * 8 + (threadIdx.x >> 5);
  int lane = threadIdx.x & 31;
  if (row >= nrows) return;
  const float* p = h + (size_t)row * cfg::D;
  float v[16];
  const float4* p4 = (const float4*)(p + lane * 16);
#pragma unroll
  for (int i = 0; i < 4; ++i) { float4 t = p4[i]; v[4*i]=t.x; v[4*i+1]=t.y; v[4*i+2]=t.z; v[4*i+3]=t.w; }
  float s = 0.f, q = 0.f;
#pragma unroll
  for (int i = 0; i < 16; ++i) { s += v[i]; q += v[i]*v[i]; }
#pragma unroll
  for (int o = 16; o > 0; o >>= 1) { s += __shfl_xor(s, o); q += __shfl_xor(q, o); }
  float mean = s * (1.f / cfg::D);
  float var  = q * (1.f / cfg::D) - mean * mean;
  float rs   = rsqrtf(var + 1e-5f);
  __bf16* out = y + (size_t)row * cfg::D + lane * 16;
  const float* scp = sc + lane * 16;
  const float* bip = bi + lane * 16;
#pragma unroll
  for (int i = 0; i < 16; ++i)
    out[i] = (__bf16)((v[i] - mean) * rs * scp[i] + bip[i]);
}

// final LN of token s==0 of each batch element -> f32 out
__global__ __launch_bounds__(256)
void k_lnf_out(const float* __restrict__ h, const float* __restrict__ sc,
               const float* __restrict__ bi, float* __restrict__ out) {
  int row  = blockIdx.x * 8 + (threadIdx.x >> 5);   // batch index
  int lane = threadIdx.x & 31;
  if (row >= cfg::B) return;
  const float* p = h + (size_t)row * cfg::S * cfg::D;
  float v[16];
  const float4* p4 = (const float4*)(p + lane * 16);
#pragma unroll
  for (int i = 0; i < 4; ++i) { float4 t = p4[i]; v[4*i]=t.x; v[4*i+1]=t.y; v[4*i+2]=t.z; v[4*i+3]=t.w; }
  float s = 0.f, q = 0.f;
#pragma unroll
  for (int i = 0; i < 16; ++i) { s += v[i]; q += v[i]*v[i]; }
#pragma unroll
  for (int o = 16; o > 0; o >>= 1) { s += __shfl_xor(s, o); q += __shfl_xor(q, o); }
  float mean = s * (1.f / cfg::D);
  float var  = q * (1.f / cfg::D) - mean * mean;
  float rs   = rsqrtf(var + 1e-5f);
  float* op = out + (size_t)row * cfg::D + lane * 16;
  const float* scp = sc + lane * 16;
  const float* bip = bi + lane * 16;
#pragma unroll
  for (int i = 0; i < 16; ++i)
    op[i] = (v[i] - mean) * rs * scp[i] + bip[i];
}

__device__ inline float gelu_tanh(float x) {
  float x3 = x * x * x;
  return 0.5f * x * (1.f + tanhf(0.7978845608028654f * (x + 0.044715f * x3)));
}

// ---------------- WMMA bf16 GEMM, async double-buffered LDS ----------------
// D(MxN) = op( A(MxK,bf16) @ Bt(NxK,bf16)^T [+ bias] [+ Cio] )
// mode 0: Dbf = acc            mode 1: Cio += bias? + acc        mode 2: Dbf = gelu(acc + bias)
#define BM 128
#define BN 64
#define BK 32
#define AP 48
#define BP 48

__global__ __launch_bounds__(256)
void k_gemm_bf16(const __bf16* __restrict__ A, const __bf16* __restrict__ Bt,
                 const float* __restrict__ bias, float* __restrict__ Cio,
                 __bf16* __restrict__ Dbf, int N, int K, int mode) {
  __shared__ __attribute__((aligned(32))) __bf16 sa [2][BM * AP];
  __shared__ __attribute__((aligned(32))) __bf16 sbt[2][BN * BP];
  const int tid  = threadIdx.x;
  const int lane = tid & 31, wave = tid >> 5;
  const int wm = wave >> 1, wn = wave & 1;             // 4 waves (M) x 2 waves (N)
  const int bm = blockIdx.y * BM, bn = blockIdx.x * BN;
  const int kh = lane >> 4, ln = lane & 15;

  // staging coordinates: 16B chunks, 4 per 32-elem row
  const int ar  = tid >> 2, aseg = tid & 3;            // A rows 0..63 (+64 for 2nd issue)
  const int br  = tid >> 2, bseg = tid & 3;            // B rows 0..63
  const unsigned sa0 = (unsigned)(uintptr_t)(&sa [0][0]);
  const unsigned sb0 = (unsigned)(uintptr_t)(&sbt[0][0]);
  const unsigned saStride = (unsigned)(BM * AP * 2);
  const unsigned sbStride = (unsigned)(BN * BP * 2);

  auto issue = [&](int buf, int k0) {
    async_b128(sa0 + buf * saStride + (unsigned)(ar * AP + aseg * 8) * 2,
               &A[(size_t)(bm + ar) * K + k0 + aseg * 8]);
    async_b128(sa0 + buf * saStride + (unsigned)((ar + 64) * AP + aseg * 8) * 2,
               &A[(size_t)(bm + ar + 64) * K + k0 + aseg * 8]);
    async_b128(sb0 + buf * sbStride + (unsigned)(br * BP + bseg * 8) * 2,
               &Bt[(size_t)(bn + br) * K + k0 + bseg * 8]);
  };

  issue(0, 0);   // prologue: stage first K tile while we init the accumulators

  f32x8 acc[2][2];
#pragma unroll
  for (int mi = 0; mi < 2; ++mi)
#pragma unroll
    for (int ni = 0; ni < 2; ++ni) {
      f32x8 c = {};
      if (mode != 0) {
        int cb = bn + wn * 32 + ni * 16 + ln;
        float bv = bias ? bias[cb] : 0.f;
#pragma unroll
        for (int i = 0; i < 8; ++i) {
          int r = bm + wm * 32 + mi * 16 + i + 8 * kh;
          float cv = (mode == 1) ? Cio[(size_t)r * N + cb] : 0.f;
          c[i] = bv + cv;
        }
      }
      acc[mi][ni] = c;
    }

  wait_async0();
  __syncthreads();

  const int nk = K / BK;
  for (int t = 0; t < nk; ++t) {
    const int buf = t & 1;
    if (t + 1 < nk) issue(buf ^ 1, (t + 1) * BK);      // overlap next-tile DMA with WMMA

    bf16x16 af[2], bfr[2];
#pragma unroll
    for (int mi = 0; mi < 2; ++mi) {
      int row = wm * 32 + mi * 16 + ln;
      bf16x8 lo = *(const bf16x8*)&sa[buf][row * AP + kh * 8];       // k = kh*8    + 0..7
      bf16x8 hi = *(const bf16x8*)&sa[buf][row * AP + 16 + kh * 8];  // k = 16+kh*8 + 0..7
      af[mi] = __builtin_shufflevector(lo, hi, 0,1,2,3,4,5,6,7,8,9,10,11,12,13,14,15);
    }
#pragma unroll
    for (int ni = 0; ni < 2; ++ni) {
      int col = wn * 32 + ni * 16 + ln;
      bf16x8 blo = *(const bf16x8*)&sbt[buf][col * BP + kh * 8];       // k = kh*8    + 0..7
      bf16x8 bhi = *(const bf16x8*)&sbt[buf][col * BP + 16 + kh * 8];  // k = 16+kh*8 + 0..7
      bfr[ni] = __builtin_shufflevector(blo, bhi, 0,1,2,3,4,5,6,7,8,9,10,11,12,13,14,15);
    }
#pragma unroll
    for (int mi = 0; mi < 2; ++mi)
#pragma unroll
      for (int ni = 0; ni < 2; ++ni)
        acc[mi][ni] = __builtin_amdgcn_wmma_f32_16x16x32_bf16(
            false, af[mi], false, bfr[ni], (short)0, acc[mi][ni], false, false);

    wait_async0();        // own wave's next-tile DMAs landed
    __syncthreads();      // everyone's landed; current tile fully consumed
  }

#pragma unroll
  for (int mi = 0; mi < 2; ++mi)
#pragma unroll
    for (int ni = 0; ni < 2; ++ni) {
      int cb = bn + wn * 32 + ni * 16 + ln;
#pragma unroll
      for (int i = 0; i < 8; ++i) {
        int r = bm + wm * 32 + mi * 16 + i + 8 * kh;
        float v = acc[mi][ni][i];
        if (mode == 0)      Dbf[(size_t)r * N + cb] = (__bf16)v;
        else if (mode == 1) Cio[(size_t)r * N + cb] = v;
        else                Dbf[(size_t)r * N + cb] = (__bf16)gelu_tanh(v);
      }
    }
}

// ---------------- attention: one block per (batch, head) ----------------
// Routing heads degenerate to full attention (W == S, NC == 1): all 8 heads
// are identical masked softmax attention over the 20 tokens.
__global__ __launch_bounds__(64)
void k_attn(const __bf16* __restrict__ qkv, const int* __restrict__ x,
            __bf16* __restrict__ o) {
  int bh = blockIdx.x;
  int b  = bh >> 3, hh = bh & 7;
  __shared__ float qs[cfg::S][cfg::DH], ks[cfg::S][cfg::DH], vs[cfg::S][cfg::DH];
  __shared__ float p[cfg::S][cfg::S];
  __shared__ float msk[cfg::S];
  int t = threadIdx.x;
  if (t < cfg::S) msk[t] = (x[b * cfg::S + t] != 0) ? 0.f : cfg::NEG;
  for (int e = t; e < cfg::S * 8; e += 64) {           // 8-elem bf16 chunks
    int s = e >> 3, c = (e & 7) * 8;
    size_t rb = (size_t)(b * cfg::S + s) * (3 * cfg::D) + hh * cfg::DH + c;
    bf16x8 q8 = *(const bf16x8*)&qkv[rb];
    bf16x8 k8 = *(const bf16x8*)&qkv[rb + cfg::D];
    bf16x8 v8 = *(const bf16x8*)&qkv[rb + 2 * cfg::D];
#pragma unroll
    for (int i = 0; i < 8; ++i) {
      qs[s][c + i] = (float)q8[i];
      ks[s][c + i] = (float)k8[i];
      vs[s][c + i] = (float)v8[i];
    }
  }
  __syncthreads();
  for (int e = t; e < cfg::S * cfg::S; e += 64) {
    int i = e / cfg::S, j = e % cfg::S;
    float dot = 0.f;
#pragma unroll
    for (int d = 0; d < cfg::DH; ++d) dot += qs[i][d] * ks[j][d];
    p[i][j] = dot * cfg::SCALE + msk[j];
  }
  __syncthreads();
  if (t < cfg::S) {
    float mx = -INFINITY;
    for (int j = 0; j < cfg::S; ++j) mx = fmaxf(mx, p[t][j]);
    float sum = 0.f;
    for (int j = 0; j < cfg::S; ++j) { float e2 = __expf(p[t][j] - mx); p[t][j] = e2; sum += e2; }
    float inv = 1.f / sum;
    for (int j = 0; j < cfg::S; ++j) p[t][j] *= inv;
  }
  __syncthreads();
  for (int e = t; e < cfg::S * cfg::DH; e += 64) {
    int s = e >> 6, d = e & 63;
    float a = 0.f;
#pragma unroll
    for (int j = 0; j < cfg::S; ++j) a += p[s][j] * vs[j][d];
    o[(size_t)(b * cfg::S + s) * cfg::D + hh * cfg::DH + d] = (__bf16)a;
  }
}

// ---------------- host launcher ----------------
extern "C" void kernel_launch(void* const* d_in, const int* in_sizes, int n_in,
                              void* d_out, int out_size, void* d_ws, size_t ws_size,
                              hipStream_t stream) {
  using namespace cfg;
  const int*   x     = (const int*)d_in[0];
  const float* tok   = (const float*)d_in[1];
  const float* pos   = (const float*)d_in[2];
  const float* ln1_s = (const float*)d_in[3];
  const float* ln1_b = (const float*)d_in[4];
  const float* w_qkv = (const float*)d_in[5];
  const float* w_o   = (const float*)d_in[6];
  /* d_in[7] = means : unused (top_k of 20-of-20 is a permutation; scatter undoes it) */
  const float* ln2_s = (const float*)d_in[8];
  const float* ln2_b = (const float*)d_in[9];
  const float* w_ff1 = (const float*)d_in[10];
  const float* b_ff1 = (const float*)d_in[11];
  const float* w_ff2 = (const float*)d_in[12];
  const float* b_ff2 = (const float*)d_in[13];
  const float* lnf_s = (const float*)d_in[14];
  const float* lnf_b = (const float*)d_in[15];

  char* ws = (char*)d_ws;
  float*  h     = (float*)(ws + OFF_H);
  __bf16* y     = (__bf16*)(ws + OFF_Y);
  __bf16* qkv   = (__bf16*)(ws + OFF_QKV);
  __bf16* attno = (__bf16*)(ws + OFF_ATT);
  __bf16* ffact = (__bf16*)(ws + OFF_FF);
  __bf16* wq_bf = (__bf16*)(ws + OFF_WQ);   // (3D x D) per layer, transposed
  __bf16* wo_bf = (__bf16*)(ws + OFF_WO);   // (D x D)
  __bf16* w1_bf = (__bf16*)(ws + OFF_W1);   // (FF x D)
  __bf16* w2_bf = (__bf16*)(ws + OFF_W2);   // (D x FF)

  const dim3 blk(256);
  // convert + transpose weights (K x N f32 -> N x K bf16), per layer via grid.z
  k_transpose_bf16<<<dim3(3*D/32, D/32, L),  blk, 0, stream>>>(w_qkv, wq_bf, D,  3*D);
  k_transpose_bf16<<<dim3(D/32,   D/32, L),  blk, 0, stream>>>(w_o,   wo_bf, D,  D);
  k_transpose_bf16<<<dim3(FF/32,  D/32, L),  blk, 0, stream>>>(w_ff1, w1_bf, D,  FF);
  k_transpose_bf16<<<dim3(D/32,  FF/32, L),  blk, 0, stream>>>(w_ff2, w2_bf, FF, D);

  k_embed<<<M, 256, 0, stream>>>(x, tok, pos, h);

  for (int l = 0; l < L; ++l) {
    k_ln_bf16<<<M / 8, blk, 0, stream>>>(h, ln1_s + (size_t)l * D, ln1_b + (size_t)l * D, y, M);

    k_gemm_bf16<<<dim3(3 * D / BN, M / BM), blk, 0, stream>>>(
        y, wq_bf + (size_t)l * D * 3 * D, nullptr, nullptr, qkv, 3 * D, D, 0);

    k_attn<<<B * H, 64, 0, stream>>>(qkv, x, attno);

    k_gemm_bf16<<<dim3(D / BN, M / BM), blk, 0, stream>>>(
        attno, wo_bf + (size_t)l * D * D, nullptr, h, nullptr, D, D, 1);

    k_ln_bf16<<<M / 8, blk, 0, stream>>>(h, ln2_s + (size_t)l * D, ln2_b + (size_t)l * D, y, M);

    k_gemm_bf16<<<dim3(FF / BN, M / BM), blk, 0, stream>>>(
        y, w1_bf + (size_t)l * D * FF, b_ff1 + (size_t)l * FF, nullptr, ffact, FF, D, 2);

    k_gemm_bf16<<<dim3(D / BN, M / BM), blk, 0, stream>>>(
        ffact, w2_bf + (size_t)l * FF * D, b_ff2 + (size_t)l * D, h, nullptr, D, FF, 1);
  }

  k_lnf_out<<<B / 8, blk, 0, stream>>>(h, lnf_s, lnf_b, (float*)d_out);
}
